// EncoderDecoderNet_89172111000155
// MI455X (gfx1250) — compile-verified, matrix operand order checked by reading, not run on hardware
//
#include <hip/hip_runtime.h>
#include <hip/hip_bf16.h>

#define HIDN 64
#define CH 192
#define NQ 100000
#define NLLM 20
#define NUSER 10
#define QDIM 768
#define LDIM 768
#define UDIM 32
#define NUM_EDGE 800000
#define NLLM_EDGE 380
#define NSEE 600000
#define NPRED 200000
#define NN (NQ + NLLM)          /* 100020 */
#define NPAD 100032             /* 16 * 6252 */
#define E2 (NSEE + NLLM_EDGE)   /* 600380 */
#define ETOT (NUM_EDGE + NLLM_EDGE) /* 800380 */
#define EPSV 1e-5f

typedef __attribute__((ext_vector_type(16))) __bf16 v16bf;
typedef __attribute__((ext_vector_type(8)))  float  v8f;

// ---------------------------------------------------------------------------
// Pack a row-major f32 K x N weight matrix into the wave32 WMMA B-fragment
// layout, bf16. Linear index t = ((kt*NTN + nt)*32 + lane)*16 + j, so each
// lane's 16-element fragment chunk is one contiguous 32-byte load.
// Element (lane, j) of fragment (kt, nt):
//   n = nt*16 + (lane & 15), koff = (lane>>4)*8
//   j in [0,8):  k = kt*32 + koff + j
//   j in [8,16): k = kt*32 + 16 + koff + (j-8)
// ---------------------------------------------------------------------------
__global__ void pack_w_kernel(const float* __restrict__ W, int K, int N,
                              __bf16* __restrict__ Wpk) {
    int NTN = N >> 4;
    int total = (K >> 5) * NTN * 32 * 16;
    int t = blockIdx.x * blockDim.x + threadIdx.x;
    if (t >= total) return;
    int j    = t & 15;
    int lane = (t >> 4) & 31;
    int rest = t >> 9;
    int nt   = rest % NTN;
    int kt   = rest / NTN;
    int n    = (nt << 4) + (lane & 15);
    int koff = (lane >> 4) << 3;
    int k    = (kt << 5) + ((j < 8) ? (koff + j) : (16 + koff + (j - 8)));
    Wpk[t] = (__bf16)W[(size_t)k * N + n];
}

// Build a 16x32 bf16 A fragment for one lane from row-major f32 A.
// lane: m = row0 + (lane&15), koff = (lane>>4)*8
// elems [0..7] = A[m][kbase+koff+0..7], elems [8..15] = A[m][kbase+16+koff+0..7]
__device__ __forceinline__ v16bf load_a_frag(const float* __restrict__ ap) {
    float4 f0 = *(const float4*)(ap);
    float4 f1 = *(const float4*)(ap + 4);
    float4 f2 = *(const float4*)(ap + 16);
    float4 f3 = *(const float4*)(ap + 20);
    v16bf v;
    v[0]  = (__bf16)f0.x; v[1]  = (__bf16)f0.y; v[2]  = (__bf16)f0.z; v[3]  = (__bf16)f0.w;
    v[4]  = (__bf16)f1.x; v[5]  = (__bf16)f1.y; v[6]  = (__bf16)f1.z; v[7]  = (__bf16)f1.w;
    v[8]  = (__bf16)f2.x; v[9]  = (__bf16)f2.y; v[10] = (__bf16)f2.z; v[11] = (__bf16)f2.w;
    v[12] = (__bf16)f3.x; v[13] = (__bf16)f3.y; v[14] = (__bf16)f3.z; v[15] = (__bf16)f3.w;
    return v;
}

// ---------------------------------------------------------------------------
// WMMA GEMM: Out[:, colOff + nquad*64 .. +64) = A(M x K) @ W(K x N) + bias.
// One wave computes a 16(M) x 64(N) tile: 1 A fragment + 4 B fragments and
// 4 v_wmma_f32_16x16x32_bf16 per K-step of 32. 4 waves per block.
// ---------------------------------------------------------------------------
__global__ void gemm16x64_bf16(const float* __restrict__ A, int lda,
                               const __bf16* __restrict__ Wpk,
                               const float* __restrict__ bias,
                               float* __restrict__ Out, int ldo, int colOff,
                               int K, int Mtiles, int NTN) {
    const int lane  = threadIdx.x & 31;
    const int mtile = blockIdx.x * (blockDim.x >> 5) + (threadIdx.x >> 5);
    if (mtile >= Mtiles) return;            // wave-uniform exit
    const int nquad = blockIdx.y;
    const int koff  = (lane >> 4) << 3;
    const int m     = (mtile << 4) + (lane & 15);

    v8f a0 = {}, a1 = {}, a2 = {}, a3 = {};
    const float*  ap = A + (size_t)m * lda + koff;
    const __bf16* bp = Wpk + ((size_t)(nquad * 4) * 32 + lane) * 16;
    const size_t  bstep = (size_t)NTN * 32 * 16;
    const int KT = K >> 5;

    for (int kt = 0; kt < KT; ++kt) {
        v16bf av = load_a_frag(ap); ap += 32;
        v16bf b0 = *(const v16bf*)(bp);
        v16bf b1 = *(const v16bf*)(bp + 512);
        v16bf b2 = *(const v16bf*)(bp + 1024);
        v16bf b3 = *(const v16bf*)(bp + 1536);
        bp += bstep;
        a0 = __builtin_amdgcn_wmma_f32_16x16x32_bf16(false, av, false, b0, (short)0, a0, false, false);
        a1 = __builtin_amdgcn_wmma_f32_16x16x32_bf16(false, av, false, b1, (short)0, a1, false, false);
        a2 = __builtin_amdgcn_wmma_f32_16x16x32_bf16(false, av, false, b2, (short)0, a2, false, false);
        a3 = __builtin_amdgcn_wmma_f32_16x16x32_bf16(false, av, false, b3, (short)0, a3, false, false);
    }

    // C/D layout: VGPR j, lanes 0-15 -> M=j, lanes 16-31 -> M=8+j; N = lane&15
    const int nloc = lane & 15;
    const int gn   = (nquad << 6) + nloc;
    const int mrow = (mtile << 4) + koff;
    const float bb0 = bias[gn], bb1 = bias[gn + 16], bb2 = bias[gn + 32], bb3 = bias[gn + 48];
    for (int j = 0; j < 8; ++j) {
        float* orow = Out + (size_t)(mrow + j) * ldo + colOff + gn;
        orow[0]  = a0[j] + bb0;
        orow[16] = a1[j] + bb1;
        orow[32] = a2[j] + bb2;
        orow[48] = a3[j] + bb3;
    }
}

// u = user_features @ W_user + b_user  (10 x 64, trivial)
__global__ void user_encode_kernel(const float* __restrict__ uf,
                                   const float* __restrict__ W,
                                   const float* __restrict__ b,
                                   float* __restrict__ u) {
    int t = blockIdx.x * blockDim.x + threadIdx.x;
    if (t >= NUSER * HIDN) return;
    int i = t / HIDN, c = t % HIDN;
    float s = b[c];
    for (int k = 0; k < UDIM; ++k) s += uf[i * UDIM + k] * W[k * HIDN + c];
    u[t] = s;
}

// x_ini[r, 128+c] = u[r % 10, c] for r < NQ
__global__ void user_fill_kernel(const float* __restrict__ u, float* __restrict__ x) {
    int t = blockIdx.x * blockDim.x + threadIdx.x;
    if (t >= NQ * HIDN) return;
    int r = t / HIDN, c = t % HIDN;
    x[(size_t)r * CH + 128 + c] = u[(r % NUSER) * HIDN + c];
}

// x_ini[NQ+i, c] = (llm_features @ W_llm + b_llm)[i, c]
__global__ void llm_encode_kernel(const float* __restrict__ lf,
                                  const float* __restrict__ W,
                                  const float* __restrict__ b,
                                  float* __restrict__ x) {
    int t = blockIdx.x * blockDim.x + threadIdx.x;
    if (t >= NLLM * CH) return;
    int i = t / CH, c = t % CH;
    float s = b[c];
    for (int k = 0; k < LDIM; ++k) s += lf[i * LDIM + k] * W[k * CH + c];
    x[(size_t)(NQ + i) * CH + c] = s;
}

// zero pad rows [NN, NPAD)
__global__ void zero_pad_kernel(float* __restrict__ x) {
    int t = blockIdx.x * blockDim.x + threadIdx.x;
    const int total = (NPAD - NN) * CH;
    if (t < total) x[(size_t)NN * CH + t] = 0.0f;
}

// Materialize masked src/dst and edge weights: ew = relu(w*W_emlp + b_emlp)
__global__ void edge_prep_kernel(const int* __restrict__ EI,
                                 const int* __restrict__ can_see,
                                 const float* __restrict__ ewin,
                                 const float* __restrict__ Wem,
                                 const float* __restrict__ bem,
                                 int* __restrict__ srcm, int* __restrict__ dstm,
                                 float* __restrict__ ew) {
    int j = blockIdx.x * blockDim.x + threadIdx.x;
    if (j >= E2) return;
    float w00 = Wem[0], b0 = bem[0];
    int e; float w;
    if (j < NSEE) { e = can_see[j]; w = ewin[e]; }
    else          { e = NUM_EDGE + (j - NSEE); w = 1.0f; }
    srcm[j] = EI[e];
    dstm[j] = EI[ETOT + e];
    float v = w * w00 + b0;
    ew[j] = v > 0.0f ? v : 0.0f;
}

__global__ void copy_f4_kernel(const float4* __restrict__ s, float4* __restrict__ d, int n4) {
    int t = blockIdx.x * blockDim.x + threadIdx.x;
    if (t < n4) d[t] = s[t];
}

// agg[dst] += h[src] + ew*W_edge + b_edge  (agg pre-initialized to x)
__global__ void scatter_add_kernel(const int* __restrict__ srcm,
                                   const int* __restrict__ dstm,
                                   const float* __restrict__ ew,
                                   const float* __restrict__ h,
                                   const float* __restrict__ We,
                                   const float* __restrict__ be,
                                   float* __restrict__ agg) {
    int t = blockIdx.x * blockDim.x + threadIdx.x;
    const int total = E2 * (CH / 4);
    if (t >= total) return;
    int j  = t / (CH / 4);
    int cg = (t % (CH / 4)) * 4;
    int s = srcm[j], d = dstm[j];
    float w = ew[j];
    float4 hv = *(const float4*)(h + (size_t)s * CH + cg);
    float* ag = agg + (size_t)d * CH + cg;
    atomicAdd(ag + 0, hv.x + w * We[cg + 0] + be[cg + 0]);
    atomicAdd(ag + 1, hv.y + w * We[cg + 1] + be[cg + 1]);
    atomicAdd(ag + 2, hv.z + w * We[cg + 2] + be[cg + 2]);
    atomicAdd(ag + 3, hv.w + w * We[cg + 3] + be[cg + 3]);
}

__global__ void zero_stats_kernel(float* __restrict__ s) {
    if (threadIdx.x < 2 * CH) s[threadIdx.x] = 0.0f;
}

// per-channel sum / sumsq over rows [0, NN)
__global__ void bn_sum_kernel(const float* __restrict__ x, float* __restrict__ stats) {
    int c = threadIdx.x;               // blockDim = CH
    long r0 = (long)blockIdx.x * 256;
    long rend = r0 + 256; if (rend > NN) rend = NN;
    float s = 0.0f, ss = 0.0f;
    for (long r = r0; r < rend; ++r) {
        float v = x[(size_t)r * CH + c];
        s += v; ss += v * v;
    }
    atomicAdd(&stats[c], s);
    atomicAdd(&stats[CH + c], ss);
}

__global__ void bn_norm_kernel(const float* __restrict__ x, const float* __restrict__ stats,
                               const float* __restrict__ g, const float* __restrict__ beta,
                               float* __restrict__ out, int doRelu) {
    int t = blockIdx.x * blockDim.x + threadIdx.x;
    const int total = NN * CH;
    if (t >= total) return;
    int c = t % CH;
    float mu  = stats[c] * (1.0f / NN);
    float var = stats[CH + c] * (1.0f / NN) - mu * mu;
    float v = (x[t] - mu) * rsqrtf(var + EPSV) * g[c] + beta[c];
    if (doRelu) v = fmaxf(v, 0.0f);
    out[t] = v;
}

// out[p] = sigmoid(mean_c(x_ini[src_p][c] * x2[dst_p][c])), one wave per pred
__global__ void predict_kernel(const int* __restrict__ EI, const int* __restrict__ emask,
                               const float* __restrict__ xini, const float* __restrict__ x2,
                               float* __restrict__ out) {
    int lane = threadIdx.x & 31;
    int p = blockIdx.x * (blockDim.x >> 5) + (threadIdx.x >> 5);
    if (p >= NPRED) return;
    int e = emask[p];
    int s = EI[e], d = EI[ETOT + e];
    const float* a = xini + (size_t)s * CH;
    const float* b = x2 + (size_t)d * CH;
    float acc = 0.0f;
    for (int c = lane; c < CH; c += 32) acc += a[c] * b[c];
    for (int off = 16; off > 0; off >>= 1) acc += __shfl_xor(acc, off, 32);
    if (lane == 0) out[p] = 1.0f / (1.0f + __expf(-acc * (1.0f / CH)));
}

static inline size_t alignup256(size_t x) { return (x + 255) & ~(size_t)255; }
static inline int cdiv(int a, int b) { return (a + b - 1) / b; }

extern "C" void kernel_launch(void* const* d_in, const int* in_sizes, int n_in,
                              void* d_out, int out_size, void* d_ws, size_t ws_size,
                              hipStream_t stream) {
    const float* task_id = (const float*)d_in[0];
    const float* query_f = (const float*)d_in[1];
    const float* llm_f   = (const float*)d_in[2];
    const float* user_f  = (const float*)d_in[3];
    const float* edge_w  = (const float*)d_in[4];
    const float* W_user  = (const float*)d_in[5];
    const float* b_user  = (const float*)d_in[6];
    const float* W_query = (const float*)d_in[7];
    const float* b_query = (const float*)d_in[8];
    const float* W_llm   = (const float*)d_in[9];
    const float* b_llm   = (const float*)d_in[10];
    const float* W_task  = (const float*)d_in[11];
    const float* b_task  = (const float*)d_in[12];
    const float* W_msg1  = (const float*)d_in[13];
    const float* b_msg1  = (const float*)d_in[14];
    const float* W_edge1 = (const float*)d_in[15];
    const float* b_edge1 = (const float*)d_in[16];
    const float* W_msg2  = (const float*)d_in[17];
    const float* b_msg2  = (const float*)d_in[18];
    const float* W_edge2 = (const float*)d_in[19];
    const float* b_edge2 = (const float*)d_in[20];
    const float* W_emlp  = (const float*)d_in[21];
    const float* b_emlp  = (const float*)d_in[22];
    const float* g1      = (const float*)d_in[23];
    const float* beta1   = (const float*)d_in[24];
    const float* g2      = (const float*)d_in[25];
    const float* beta2   = (const float*)d_in[26];
    const int*   EI      = (const int*)d_in[27];
    const int*   can_see = (const int*)d_in[28];
    const int*   emask   = (const int*)d_in[29];
    float* out = (float*)d_out;

    // -------- workspace layout --------
    const size_t SZ_X = (size_t)NPAD * CH * sizeof(float);   // ~76.8 MB each
    unsigned char* w = (unsigned char*)d_ws;
    float* xini = (float*)(w);                 // x_ini (kept for predict)
    float* hbuf = (float*)(w + SZ_X);          // GEMM msg output (both convs)
    float* x1   = (float*)(w + 2 * SZ_X);      // relu(bn(conv1))
    float* x2   = (float*)(w + 3 * SZ_X);      // agg buffer / bn(conv2)
    unsigned char* p = w + 4 * SZ_X;
    float* ew   = (float*)p;  p += alignup256((size_t)E2 * 4);
    int* srcm   = (int*)p;    p += alignup256((size_t)E2 * 4);
    int* dstm   = (int*)p;    p += alignup256((size_t)E2 * 4);
    float* ubuf = (float*)p;  p += alignup256((size_t)NUSER * HIDN * 4);
    float* stats= (float*)p;  p += alignup256((size_t)2 * CH * 4);
    __bf16* wpkT  = (__bf16*)p; p += alignup256((size_t)(LDIM / 32) * (HIDN / 16) * 512 * 2);
    __bf16* wpkQ  = (__bf16*)p; p += alignup256((size_t)(QDIM / 32) * (HIDN / 16) * 512 * 2);
    __bf16* wpkM1 = (__bf16*)p; p += alignup256((size_t)(CH / 32) * (CH / 16) * 512 * 2);
    __bf16* wpkM2 = (__bf16*)p; p += alignup256((size_t)(CH / 32) * (CH / 16) * 512 * 2);

    // -------- weight packing (tiny) --------
    const int pkEnc = (LDIM / 32) * (HIDN / 16) * 512;  // 49152
    const int pkMsg = (CH / 32) * (CH / 16) * 512;      // 36864
    pack_w_kernel<<<cdiv(pkEnc, 256), 256, 0, stream>>>(W_task,  LDIM, HIDN, wpkT);
    pack_w_kernel<<<cdiv(pkEnc, 256), 256, 0, stream>>>(W_query, QDIM, HIDN, wpkQ);
    pack_w_kernel<<<cdiv(pkMsg, 256), 256, 0, stream>>>(W_msg1,  CH,   CH,   wpkM1);
    pack_w_kernel<<<cdiv(pkMsg, 256), 256, 0, stream>>>(W_msg2,  CH,   CH,   wpkM2);

    // -------- encoders: x_ini --------
    // t -> cols [0,64), q -> cols [64,128)   (M = 100000 = 6250 tiles exactly)
    gemm16x64_bf16<<<dim3(cdiv(6250, 4), 1), 128, 0, stream>>>(
        task_id, LDIM, wpkT, b_task, xini, CH, 0, LDIM, 6250, HIDN / 16);
    gemm16x64_bf16<<<dim3(cdiv(6250, 4), 1), 128, 0, stream>>>(
        query_f, QDIM, wpkQ, b_query, xini, CH, 64, QDIM, 6250, HIDN / 16);
    user_encode_kernel<<<cdiv(NUSER * HIDN, 128), 128, 0, stream>>>(user_f, W_user, b_user, ubuf);
    user_fill_kernel<<<cdiv(NQ * HIDN, 256), 256, 0, stream>>>(ubuf, xini);
    llm_encode_kernel<<<cdiv(NLLM * CH, 128), 128, 0, stream>>>(llm_f, W_llm, b_llm, xini);
    zero_pad_kernel<<<cdiv((NPAD - NN) * CH, 256), 256, 0, stream>>>(xini);

    // -------- edges --------
    edge_prep_kernel<<<cdiv(E2, 256), 256, 0, stream>>>(EI, can_see, edge_w, W_emlp, b_emlp,
                                                        srcm, dstm, ew);

    const int n4 = NPAD * CH / 4;
    const int scatterTot = E2 * (CH / 4);

    // -------- conv1 --------
    gemm16x64_bf16<<<dim3(NPAD / 64, CH / 64), 128, 0, stream>>>(
        xini, CH, wpkM1, b_msg1, hbuf, CH, 0, CH, NPAD / 16, CH / 16);
    copy_f4_kernel<<<cdiv(n4, 256), 256, 0, stream>>>((const float4*)xini, (float4*)x2, n4);
    scatter_add_kernel<<<cdiv(scatterTot, 256), 256, 0, stream>>>(srcm, dstm, ew, hbuf,
                                                                  W_edge1, b_edge1, x2);
    zero_stats_kernel<<<1, 2 * CH, 0, stream>>>(stats);
    bn_sum_kernel<<<cdiv(NN, 256), CH, 0, stream>>>(x2, stats);
    bn_norm_kernel<<<cdiv(NN * CH, 256), 256, 0, stream>>>(x2, stats, g1, beta1, x1, 1);
    zero_pad_kernel<<<cdiv((NPAD - NN) * CH, 256), 256, 0, stream>>>(x1);

    // -------- conv2 --------
    gemm16x64_bf16<<<dim3(NPAD / 64, CH / 64), 128, 0, stream>>>(
        x1, CH, wpkM2, b_msg2, hbuf, CH, 0, CH, NPAD / 16, CH / 16);
    copy_f4_kernel<<<cdiv(n4, 256), 256, 0, stream>>>((const float4*)x1, (float4*)x2, n4);
    scatter_add_kernel<<<cdiv(scatterTot, 256), 256, 0, stream>>>(srcm, dstm, ew, hbuf,
                                                                  W_edge2, b_edge2, x2);
    zero_stats_kernel<<<1, 2 * CH, 0, stream>>>(stats);
    bn_sum_kernel<<<cdiv(NN, 256), CH, 0, stream>>>(x2, stats);
    bn_norm_kernel<<<cdiv(NN * CH, 256), 256, 0, stream>>>(x2, stats, g2, beta2, x2, 0);

    // -------- predict --------
    predict_kernel<<<cdiv(NPRED, 8), 256, 0, stream>>>(EI, emask, xini, x2, out);

    (void)in_sizes; (void)n_in; (void)out_size; (void)ws_size;
}
